// DeltaNetPlus_42185168781699
// MI455X (gfx1250) — matrix operand order repaired and checked by reading, not compiled, not run on hardware
//
#include <hip/hip_runtime.h>

// ---------------------------------------------------------------------------
// CDNA5 (gfx1250) DeltaNet-Plus forward — bf16-resident dataflow.
// wave32; WMMA bf16 16x16x32 (fp32 accum) for every dense matmul.
// GEMM uses async global->LDS staging (global_load_async_to_lds_b128,
// ASYNCcnt) + speculative global_prefetch_b8. The 256x256 recurrent state
// lives in LDS for the whole 64-chunk recurrence.
// ---------------------------------------------------------------------------

typedef __attribute__((ext_vector_type(16))) __bf16       v16bf;
typedef __attribute__((ext_vector_type(8)))  float        v8f;
typedef __attribute__((ext_vector_type(4)))  float        v4f;
typedef __attribute__((ext_vector_type(4)))  unsigned int v4u;

union FragBF { v16bf v; v4u q[2]; unsigned short u[16]; };
union H8     { v4u q; unsigned short u[8]; };

__device__ __forceinline__ unsigned short f2bf(float f) {
  unsigned int x = __float_as_uint(f);
  unsigned int r = x + 0x7FFFu + ((x >> 16) & 1u);   // round-to-nearest-even
  return (unsigned short)(r >> 16);
}
__device__ __forceinline__ float bf2f(unsigned short s) {
  return __uint_as_float(((unsigned int)s) << 16);
}
__device__ __forceinline__ v8f zero_v8f() {
  v8f z = {0.f, 0.f, 0.f, 0.f, 0.f, 0.f, 0.f, 0.f};
  return z;
}
__device__ __forceinline__ v8f wmma_bf16(v16bf a, v16bf b, v8f c) {
  return __builtin_amdgcn_wmma_f32_16x16x32_bf16(false, a, false, b, (short)0, c,
                                                 false, false);
}

// ---- CDNA5 async global->LDS (VGLOBAL encoding, tracked by ASYNCcnt) ------
// LDS address is taken from the VDST VGPR; low 32 bits of a shared-aperture generic
// address are the LDS byte offset (ISA 10.2).
__device__ __forceinline__ void async_b128_to_lds(void* lds_ptr, const void* gptr) {
  unsigned loff = (unsigned)(size_t)lds_ptr;
  asm volatile("global_load_async_to_lds_b128 %0, %1, off"
               :: "v"(loff), "v"(gptr) : "memory");
}
__device__ __forceinline__ void wait_asynccnt0() {
  asm volatile("s_wait_asynccnt 0x0" ::: "memory");
}

// ---- fragment builders (ISA 7.12.2 wave32 layouts), all 16B vector loads --
// A operand (16x32): lane row M=lane&15; e<8 -> K=kb+e, e>=8 -> K=16+kb+e-8,
//   kb=(lane>>4)*8  => two 16B groups at [kb] and [kb+16].
// B operand (32x16): lane col N=lane&15; same K mapping as A (kb=(lane>>4)*8)
//   => two 16B groups at [kb] and [kb+16].
// C/D (16x16 f32): vgpr v -> M=v+8*(lane>>4), N=lane&15.

__device__ __forceinline__ v16bf frag_a_lds(const unsigned short* X, int ld,
                                            int row0, int k0, int lane) {
  FragBF f;
  const unsigned short* p = X + (row0 + (lane & 15)) * ld + k0 + ((lane >> 4) << 3);
  f.q[0] = *(const v4u*)p;
  f.q[1] = *(const v4u*)(p + 16);
  return f.v;
}
__device__ __forceinline__ v16bf frag_a_lds_neg(const unsigned short* X, int ld,
                                                int row0, int k0, int lane) {
  FragBF f;
  const unsigned short* p = X + (row0 + (lane & 15)) * ld + k0 + ((lane >> 4) << 3);
  f.q[0] = (*(const v4u*)p) ^ 0x80008000u;        // bf16 sign-flip
  f.q[1] = (*(const v4u*)(p + 16)) ^ 0x80008000u;
  return f.v;
}
__device__ __forceinline__ v16bf frag_a_gbl_bf(const unsigned short* __restrict__ X,
                                               size_t ld, size_t row0, size_t k0,
                                               int lane) {
  FragBF f;
  const unsigned short* p = X + (row0 + (size_t)(lane & 15)) * ld + k0 +
                            (size_t)((lane >> 4) << 3);
  f.q[0] = *(const v4u*)p;
  f.q[1] = *(const v4u*)(p + 16);
  return f.v;
}
// A-frag from fp32 LDS rows with per-column scale (Tb = T * diag(beta))
__device__ __forceinline__ v16bf frag_a_f32_colscaled(const float* X, int ld,
                                                      int row0, int k0,
                                                      const float* cs, int lane) {
  FragBF f;
  int kb = (lane >> 4) << 3;
  const float* p = X + (row0 + (lane & 15)) * ld + k0 + kb;
  const float* s = cs + k0 + kb;
  v4f a0 = *(const v4f*)p,        a1 = *(const v4f*)(p + 4);
  v4f a2 = *(const v4f*)(p + 16), a3 = *(const v4f*)(p + 20);
  v4f s0 = *(const v4f*)s,        s1 = *(const v4f*)(s + 4);
  v4f s2 = *(const v4f*)(s + 16), s3 = *(const v4f*)(s + 20);
#pragma unroll
  for (int e = 0; e < 4; ++e) {
    f.u[e] = f2bf(a0[e] * s0[e]); f.u[4 + e] = f2bf(a1[e] * s1[e]);
    f.u[8 + e] = f2bf(a2[e] * s2[e]); f.u[12 + e] = f2bf(a3[e] * s3[e]);
  }
  return f.v;
}
__device__ __forceinline__ v16bf frag_bT_lds(const unsigned short* Y, int ld,
                                             int k0, int col0, int lane) {
  FragBF f;
  const unsigned short* p = Y + (col0 + (lane & 15)) * ld + k0 + ((lane >> 4) << 3);
  f.q[0] = *(const v4u*)p;
  f.q[1] = *(const v4u*)(p + 16);
  return f.v;
}
__device__ __forceinline__ v16bf frag_bT_gbl_bf(const unsigned short* __restrict__ Y,
                                                size_t ld, size_t rowbase, size_t k0,
                                                int col0, int lane) {
  FragBF f;
  const unsigned short* p = Y + (rowbase + (size_t)(col0 + (lane & 15))) * ld + k0 +
                            (size_t)((lane >> 4) << 3);
  f.q[0] = *(const v4u*)p;
  f.q[1] = *(const v4u*)(p + 16);
  return f.v;
}

__device__ __forceinline__ void store_acc_row_bf16(unsigned short* X, int ld,
                                                   int row0, int col0, v8f acc,
                                                   int lane) {
  int mb = (lane >> 4) << 3, n = lane & 15;
#pragma unroll
  for (int el = 0; el < 8; ++el)
    X[(row0 + el + mb) * ld + col0 + n] = f2bf(acc[el]);
}
__device__ __forceinline__ void store_acc_colT_bf16(unsigned short* X, int ld,
                                                    int row0, int col0, v8f acc,
                                                    int lane) {
  H8 h;
#pragma unroll
  for (int el = 0; el < 8; ++el) h.u[el] = f2bf(acc[el]);
  *(v4u*)&X[(col0 + (lane & 15)) * ld + row0 + ((lane >> 4) << 3)] = h.q;
}

// ---------------------------------------------------------------------------
__global__ void cvt_f32_to_bf16(const float* __restrict__ in,
                                unsigned short* __restrict__ out, size_t n) {
  for (size_t i = ((size_t)blockIdx.x * blockDim.x + threadIdx.x) * 8; i < n;
       i += (size_t)gridDim.x * blockDim.x * 8) {
    v4f a = *(const v4f*)&in[i];
    v4f b = *(const v4f*)&in[i + 4];
    H8 h;
#pragma unroll
    for (int e = 0; e < 4; ++e) { h.u[e] = f2bf(a[e]); h.u[4 + e] = f2bf(b[e]); }
    *(v4u*)&out[i] = h.q;
  }
}

// ---------------------------------------------------------------------------
// GEMM: C[M,N] = A[M,K] * B[K,N], A/B bf16 global, fp32 accum/out.
// 256 threads = 8 waves; 128x64 block tile; each wave owns a 16x64 strip:
// one shared A-fragment feeds 4 WMMAs per K-step. A tile staged with
// async global->LDS b128; B tile transposed manually (needs 2B scatter).
// ---------------------------------------------------------------------------
__global__ __launch_bounds__(256)
void gemm_bf16_wmma(const unsigned short* __restrict__ A,
                    const unsigned short* __restrict__ B,
                    float* __restrict__ C, int M, int N, int K) {
  __shared__ unsigned short sA[128 * 40];   // [m][k]
  __shared__ unsigned short sBT[64 * 40];   // [n][k]
  const int tid = threadIdx.x, lane = tid & 31, w = tid >> 5;   // w: 0..7
  const int m0 = blockIdx.y * 128, n0 = blockIdx.x * 64;

  v8f acc[4] = {zero_v8f(), zero_v8f(), zero_v8f(), zero_v8f()};

  for (int k0 = 0; k0 < K; k0 += 32) {
    __syncthreads();
    {   // A tile 128x32: two async 16B global->LDS copies per thread
      int r = tid >> 2, c8 = (tid & 3) * 8;
      async_b128_to_lds(&sA[r * 40 + c8],
                        &A[(size_t)(m0 + r) * K + k0 + c8]);
      async_b128_to_lds(&sA[(r + 64) * 40 + c8],
                        &A[(size_t)(m0 + r + 64) * K + k0 + c8]);
    }
    {   // B tile 32x64 -> transposed sBT[n][k]
      int kk = tid & 31, n8 = (tid >> 5) * 8;
      H8 hb;
      hb.q = *(const v4u*)&B[(size_t)(k0 + kk) * N + n0 + n8];
#pragma unroll
      for (int e = 0; e < 8; ++e) sBT[(n8 + e) * 40 + kk] = hb.u[e];
    }
    // speculative prefetch of the next K tile (dropped if OOB)
    __builtin_prefetch(&A[(size_t)(m0 + (tid >> 2)) * K + k0 + 32], 0, 1);
    __builtin_prefetch(&B[(size_t)(k0 + 32 + (tid & 31)) * N + n0], 0, 1);
    wait_asynccnt0();
    __syncthreads();

    v16bf af = frag_a_lds(sA, 40, 16 * w, 0, lane);
#pragma unroll
    for (int t = 0; t < 4; ++t)
      acc[t] = wmma_bf16(af, frag_bT_lds(sBT, 40, 0, 16 * t, lane), acc[t]);
  }

  const int mb = (lane >> 4) << 3, nn = lane & 15;
#pragma unroll
  for (int t = 0; t < 4; ++t) {
#pragma unroll
    for (int el = 0; el < 8; ++el) {
      size_t row = (size_t)(m0 + 16 * w + el + mb);
      C[row * N + n0 + 16 * t + nn] = acc[t][el];
    }
  }
}

// ---------------------------------------------------------------------------
__global__ void beta_kernel(const float* __restrict__ x, const float* __restrict__ Wb,
                            float* __restrict__ beta, int BL) {
  int t = blockIdx.x * blockDim.x + threadIdx.x;
  if (t >= BL * 4) return;
  int pos = t >> 2, h = t & 3;
  const float* xr = x + (size_t)pos * 1024;
  float acc = 0.f;
  for (int i = 0; i < 1024; i += 4) {
    v4f xv = *(const v4f*)&xr[i];
#pragma unroll
    for (int e = 0; e < 4; ++e) acc += xv[e] * Wb[(i + e) * 4 + h];
  }
  beta[t] = 1.f / (1.f + __expf(-acc));
}

__global__ void conv_silu_res_f32(const float* __restrict__ pre,
                                  const float* __restrict__ w,
                                  float* __restrict__ post, int B, int L) {
  size_t total = (size_t)B * L * 1024;
  for (size_t idx = (size_t)blockIdx.x * blockDim.x + threadIdx.x; idx < total;
       idx += (size_t)gridDim.x * blockDim.x) {
    int c = (int)(idx & 1023);
    size_t bl = idx >> 10;
    int l = (int)(bl % L);
    size_t b = bl / L;
    float y = 0.f;
#pragma unroll
    for (int j = 0; j < 4; ++j) {
      int ls = l - 3 + j;
      if (ls >= 0) y += w[c * 4 + j] * pre[((b * L + (size_t)ls) << 10) + c];
    }
    float s = y / (1.f + __expf(-y));
    post[idx] = s + pre[idx];
  }
}
__global__ void conv_silu_res_bf16(const float* __restrict__ pre,
                                   const float* __restrict__ w,
                                   unsigned short* __restrict__ post, int B, int L) {
  size_t total = (size_t)B * L * 1024;
  for (size_t idx = (size_t)blockIdx.x * blockDim.x + threadIdx.x; idx < total;
       idx += (size_t)gridDim.x * blockDim.x) {
    int c = (int)(idx & 1023);
    size_t bl = idx >> 10;
    int l = (int)(bl % L);
    size_t b = bl / L;
    float y = 0.f;
#pragma unroll
    for (int j = 0; j < 4; ++j) {
      int ls = l - 3 + j;
      if (ls >= 0) y += w[c * 4 + j] * pre[((b * L + (size_t)ls) << 10) + c];
    }
    float s = y / (1.f + __expf(-y));
    post[idx] = f2bf(s + pre[idx]);
  }
}

__global__ void l2norm_rows_bf16(const float* __restrict__ in,
                                 unsigned short* __restrict__ out, int nrows,
                                 float scale) {
  int lane = threadIdx.x & 31;
  int row = blockIdx.x * (blockDim.x >> 5) + (threadIdx.x >> 5);
  if (row >= nrows) return;
  const float* p = in + (size_t)row * 256 + lane * 8;
  v4f v0 = *(const v4f*)p, v1 = *(const v4f*)(p + 4);
  float ss = 0.f;
#pragma unroll
  for (int e = 0; e < 4; ++e) ss += v0[e] * v0[e] + v1[e] * v1[e];
#pragma unroll
  for (int o = 16; o > 0; o >>= 1) ss += __shfl_xor(ss, o, 32);
  float inv = rsqrtf(ss) * scale;
  H8 h;
#pragma unroll
  for (int e = 0; e < 4; ++e) { h.u[e] = f2bf(v0[e] * inv); h.u[4 + e] = f2bf(v1[e] * inv); }
  *(v4u*)&out[(size_t)row * 256 + lane * 8] = h.q;
}

__global__ void rmsnorm_rows_bf16(const float* __restrict__ o,
                                  const float* __restrict__ nw,
                                  unsigned short* __restrict__ out, int nrows) {
  int lane = threadIdx.x & 31;
  int row = blockIdx.x * (blockDim.x >> 5) + (threadIdx.x >> 5);
  if (row >= nrows) return;
  const float* p = o + (size_t)row * 256 + lane * 8;
  v4f v0 = *(const v4f*)p, v1 = *(const v4f*)(p + 4);
  float ss = 0.f;
#pragma unroll
  for (int e = 0; e < 4; ++e) ss += v0[e] * v0[e] + v1[e] * v1[e];
#pragma unroll
  for (int oo = 16; oo > 0; oo >>= 1) ss += __shfl_xor(ss, oo, 32);
  float inv = rsqrtf(ss * (1.f / 256.f) + 1e-5f);
  v4f w0 = *(const v4f*)&nw[lane * 8], w1 = *(const v4f*)&nw[lane * 8 + 4];
  H8 h;
#pragma unroll
  for (int e = 0; e < 4; ++e) {
    h.u[e] = f2bf(v0[e] * inv * w0[e]);
    h.u[4 + e] = f2bf(v1[e] * inv * w1[e]);
  }
  *(v4u*)&out[(size_t)row * 256 + lane * 8] = h.q;
}

// ---------------------------------------------------------------------------
// Chunked delta-rule scan. One block per (b,h); 512 threads = 16 waves.
// LDS 323,840 B of the 320KB WGP. See round-3 layout comments.
// ---------------------------------------------------------------------------
#define LDS_S 264
#define LDT   72
#define LDA   68

__global__ __launch_bounds__(512)
void deltanet_scan(const unsigned short* __restrict__ qg,
                   const unsigned short* __restrict__ kg,
                   const unsigned short* __restrict__ vg,
                   const float* __restrict__ betag, const float* __restrict__ S0,
                   float* __restrict__ og, float* __restrict__ Sout, int B, int L) {
  extern __shared__ char smem[];
  unsigned short* sS2  = (unsigned short*)smem;     // 256*264
  unsigned short* sKt  = sS2 + 256 * LDS_S;         // 256*72
  unsigned short* sVt  = sKt + 256 * LDT;           // 256*72
  unsigned short* sUWt = sVt + 256 * LDT;           // 256*72
  unsigned short* sW   = sUWt + 256 * LDT;          // 64*264
  unsigned short* sP   = sW + 64 * LDS_S;           // 64*72
  float* sA    = (float*)(sP + 64 * LDT);           // 64*68
  float* sT    = sA + 64 * LDA;                     // 64*68
  float* sbeta = sT + 64 * LDA;                     // 64

  const int tid = threadIdx.x, lane = tid & 31, w = tid >> 5;  // w: 0..15
  const int bh = blockIdx.x;
  const int b = bh >> 2, h = bh & 3;
  const size_t headoff = (size_t)h * 256;
  const int mb = (lane >> 4) << 3, nn = lane & 15;

  for (int i = tid; i < 256 * 256; i += 512) {
    int dv = i >> 8, dk = i & 255;
    sS2[dv * LDS_S + dk] = f2bf(S0[((size_t)bh << 16) + i]);
  }

  const int Nc = L / 64;
  for (int c = 0; c < Nc; ++c) {
    const size_t rowbase = (size_t)(b * L + c * 64);
    __syncthreads();
    if (tid < 64) sbeta[tid] = betag[(rowbase + tid) * 4 + h];
    for (int i = tid; i < 64 * 32; i += 512) {
      int r = i >> 5, d8 = (i & 31) * 8;
      size_t g = ((rowbase + r) << 10) + headoff + d8;
      H8 hk, hv;
      hk.q = *(const v4u*)&kg[g];
      hv.q = *(const v4u*)&vg[g];
#pragma unroll
      for (int e = 0; e < 8; ++e) {
        sKt[(d8 + e) * LDT + r] = hk.u[e];
        sVt[(d8 + e) * LDT + r] = hv.u[e];
      }
    }
    __syncthreads();

    // ---- Phase A: A = I + tril(diag(beta) * K K^T, -1) ----
    {
      int ti = w >> 2, tj = w & 3;
      v8f acc = zero_v8f();
      for (int ks = 0; ks < 8; ++ks) {
        v16bf a  = frag_a_gbl_bf(kg, 1024, rowbase + 16 * ti,
                                 headoff + (size_t)(ks * 32), lane);
        v16bf bb = frag_bT_gbl_bf(kg, 1024, rowbase,
                                  headoff + (size_t)(ks * 32), 16 * tj, lane);
        acc = wmma_bf16(a, bb, acc);
      }
#pragma unroll
      for (int el = 0; el < 8; ++el) {
        int gi = 16 * ti + el + mb, gj = 16 * tj + nn;
        sA[gi * LDA + gj] = (gj < gi) ? acc[el] * sbeta[gi]
                                      : (gi == gj ? 1.0f : 0.0f);
      }
    }
    __syncthreads();

    // ---- Phase C: T = A^{-1} (unit lower), per-column forward substitution ----
    if (tid < 64) {
      int j = tid;
      for (int i = 0; i < 64; ++i) {
        float s = (i == j) ? 1.f : 0.f;
        for (int p = j; p < i; ++p) s -= sA[i * LDA + p] * sT[p * LDA + j];
        sT[i * LDA + j] = s;
      }
    }
    __syncthreads();

    // ---- Phases D1+D2: W = Tb@K ; UW = Tb@V - W@S^T ----
    {
      int ti = w & 3, tjg = w >> 2;
      v16bf tb0 = frag_a_f32_colscaled(sT, LDA, 16 * ti, 0, sbeta, lane);
      v16bf tb1 = frag_a_f32_colscaled(sT, LDA, 16 * ti, 32, sbeta, lane);
#pragma unroll
      for (int t = 0; t < 4; ++t) {
        int tj = tjg * 4 + t;
        v8f a = zero_v8f();
        a = wmma_bf16(tb0, frag_bT_lds(sKt, LDT, 0, 16 * tj, lane), a);
        a = wmma_bf16(tb1, frag_bT_lds(sKt, LDT, 32, 16 * tj, lane), a);
        store_acc_row_bf16(sW, LDS_S, 16 * ti, 16 * tj, a, lane);
      }
      __syncthreads();
      v8f accU[4];
#pragma unroll
      for (int t = 0; t < 4; ++t) {
        int tj = tjg * 4 + t;
        v8f a = zero_v8f();
        a = wmma_bf16(tb0, frag_bT_lds(sVt, LDT, 0, 16 * tj, lane), a);
        a = wmma_bf16(tb1, frag_bT_lds(sVt, LDT, 32, 16 * tj, lane), a);
        accU[t] = a;
      }
      for (int ks = 0; ks < 8; ++ks) {
        v16bf aW = frag_a_lds_neg(sW, LDS_S, 16 * ti, ks * 32, lane);   // -W
#pragma unroll
        for (int t = 0; t < 4; ++t)
          accU[t] = wmma_bf16(aW, frag_bT_lds(sS2, LDS_S, ks * 32,
                                              16 * (tjg * 4 + t), lane), accU[t]);
      }
#pragma unroll
      for (int t = 0; t < 4; ++t)
        store_acc_colT_bf16(sUWt, LDT, 16 * ti, 16 * (tjg * 4 + t), accU[t], lane);
    }
    __syncthreads();

    // ---- Phase D3: P = (Q K^T) ∘ lower-mask (waves 0..3) ----
    if (w < 4) {
      int ti = w;
      v8f acc[4] = {zero_v8f(), zero_v8f(), zero_v8f(), zero_v8f()};
      for (int ks = 0; ks < 8; ++ks) {
        v16bf a = frag_a_gbl_bf(qg, 1024, rowbase + 16 * ti,
                                headoff + (size_t)(ks * 32), lane);
#pragma unroll
        for (int t = 0; t < 4; ++t)
          acc[t] = wmma_bf16(a, frag_bT_gbl_bf(kg, 1024, rowbase,
                                               headoff + (size_t)(ks * 32),
                                               16 * t, lane), acc[t]);
      }
#pragma unroll
      for (int t = 0; t < 4; ++t) {
#pragma unroll
        for (int el = 0; el < 8; ++el) {
          int gi = 16 * ti + el + mb, gj = 16 * t + nn;
          sP[gi * LDT + gj] = (gj <= gi) ? f2bf(acc[t][el]) : (unsigned short)0;
        }
      }
    }
    __syncthreads();

    // ---- Phase E: O = Q @ S^T + P @ UW -> global ----
    {
      int ti = w & 3, tjg = w >> 2;
      v8f acc[4] = {zero_v8f(), zero_v8f(), zero_v8f(), zero_v8f()};
      for (int ks = 0; ks < 8; ++ks) {
        v16bf a = frag_a_gbl_bf(qg, 1024, rowbase + 16 * ti,
                                headoff + (size_t)(ks * 32), lane);
#pragma unroll
        for (int t = 0; t < 4; ++t)
          acc[t] = wmma_bf16(a, frag_bT_lds(sS2, LDS_S, ks * 32,
                                            16 * (tjg * 4 + t), lane), acc[t]);
      }
      for (int ks = 0; ks < 2; ++ks) {
        v16bf a = frag_a_lds(sP, LDT, 16 * ti, ks * 32, lane);
#pragma unroll
        for (int t = 0; t < 4; ++t)
          acc[t] = wmma_bf16(a, frag_bT_lds(sUWt, LDT, ks * 32,
                                            16 * (tjg * 4 + t), lane), acc[t]);
      }
#pragma unroll
      for (int t = 0; t < 4; ++t) {
#pragma unroll
        for (int el = 0; el < 8; ++el) {
          size_t g = ((rowbase + 16 * ti + el + mb) << 10) + headoff +
                     16 * (tjg * 4 + t) + nn;
          og[g] = acc[t][el];
        }
      }
    }
    __syncthreads();

    // ---- Phase F: S^T += K^T @ UW ----
    {
      v16bf ka0 = frag_a_lds(sKt, LDT, 16 * w, 0, lane);
      v16bf ka1 = frag_a_lds(sKt, LDT, 16 * w, 32, lane);
      for (int tj = 0; tj < 16; ++tj) {
        int soff = (16 * tj + nn) * LDS_S + 16 * w + mb;
        H8 hl;
        hl.q = *(const v4u*)&sS2[soff];
        v8f acc;
#pragma unroll
        for (int el = 0; el < 8; ++el) acc[el] = bf2f(hl.u[el]);
        acc = wmma_bf16(ka0, frag_bT_lds(sUWt, LDT, 0, 16 * tj, lane), acc);
        acc = wmma_bf16(ka1, frag_bT_lds(sUWt, LDT, 32, 16 * tj, lane), acc);
        H8 hs;
#pragma unroll
        for (int el = 0; el < 8; ++el) hs.u[el] = f2bf(acc[el]);
        *(v4u*)&sS2[soff] = hs.q;
      }
    }
  }

  __syncthreads();
  for (int i = tid; i < 256 * 256; i += 512) {
    int dv = i >> 8, dk = i & 255;
    Sout[((size_t)bh << 16) + i] = bf2f(sS2[dv * LDS_S + dk]);
  }
}

// ---------------------------------------------------------------------------
extern "C" void kernel_launch(void* const* d_in, const int* in_sizes, int n_in,
                              void* d_out, int out_size, void* d_ws, size_t ws_size,
                              hipStream_t stream) {
  (void)in_sizes; (void)n_in; (void)out_size; (void)ws_size;

  const float* x  = (const float*)d_in[0];
  const float* Wq = (const float*)d_in[1];
  const float* Wk = (const float*)d_in[2];
  const float* Wv = (const float*)d_in[3];
  const float* Wb = (const float*)d_in[4];
  const float* cq = (const float*)d_in[5];
  const float* ck = (const float*)d_in[6];
  const float* cv = (const float*)d_in[7];
  const float* nw = (const float*)d_in[8];
  const float* Wo = (const float*)d_in[9];
  const float* S0 = (const float*)d_in[10];

  const int B = 2, L = 4096, HIDc = 1024;
  const size_t BL = (size_t)B * L;         // 8192
  const size_t MAT = BL * HIDc;            // 8,388,608
  const size_t WSZ = (size_t)HIDc * HIDc;  // 1,048,576

  float* ws = (float*)d_ws;
  float* q_pre  = ws;                      // [MAT] f32
  float* k_pre  = ws + MAT;                // [MAT] f32
  float* v_pre  = ws + 2 * MAT;            // [MAT] f32
  float* q_post = ws + 3 * MAT;            // [MAT] f32
  float* k_post = ws + 4 * MAT;            // [MAT] f32
  float* beta   = ws + 5 * MAT;            // [BL*4] f32
  unsigned short* x_bf  = (unsigned short*)(ws + 5 * MAT + 32768);  // MAT halves
  unsigned short* Wq_bf = x_bf + MAT;
  unsigned short* Wk_bf = Wq_bf + WSZ;
  unsigned short* Wv_bf = Wk_bf + WSZ;
  unsigned short* Wo_bf = Wv_bf + WSZ;
  unsigned short* q_bf  = (unsigned short*)q_pre;   // after conv-q
  unsigned short* k_bf  = q_bf + MAT;               // (2nd half of q_pre)
  unsigned short* v_bf  = (unsigned short*)k_pre;   // after conv-k
  unsigned short* o_nrm = v_bf + MAT;               // (2nd half of k_pre)
  float*          o_raw = v_pre;                    // after conv-v

  float* out  = (float*)d_out;             // [B,L,HID]
  float* Sout = out + MAT;                 // [B,H,256,256]

  dim3 gblk(256);
  dim3 ggrid(HIDc / 64, (unsigned)(BL / 128));   // (16, 64)

  cvt_f32_to_bf16<<<2048, 256, 0, stream>>>(x,  x_bf,  MAT);
  cvt_f32_to_bf16<<<512, 256, 0, stream>>>(Wq, Wq_bf, WSZ);
  cvt_f32_to_bf16<<<512, 256, 0, stream>>>(Wk, Wk_bf, WSZ);
  cvt_f32_to_bf16<<<512, 256, 0, stream>>>(Wv, Wv_bf, WSZ);
  cvt_f32_to_bf16<<<512, 256, 0, stream>>>(Wo, Wo_bf, WSZ);

  gemm_bf16_wmma<<<ggrid, gblk, 0, stream>>>(x_bf, Wq_bf, q_pre, (int)BL, HIDc, HIDc);
  gemm_bf16_wmma<<<ggrid, gblk, 0, stream>>>(x_bf, Wk_bf, k_pre, (int)BL, HIDc, HIDc);
  gemm_bf16_wmma<<<ggrid, gblk, 0, stream>>>(x_bf, Wv_bf, v_pre, (int)BL, HIDc, HIDc);

  beta_kernel<<<(unsigned)((BL * 4 + 255) / 256), 256, 0, stream>>>(x, Wb, beta, (int)BL);

  conv_silu_res_f32<<<2048, 256, 0, stream>>>(q_pre, cq, q_post, B, L);
  conv_silu_res_f32<<<2048, 256, 0, stream>>>(k_pre, ck, k_post, B, L);
  conv_silu_res_bf16<<<2048, 256, 0, stream>>>(v_pre, cv, v_bf, B, L);

  const int rows = (int)(BL * 4);          // 32768 head-rows of 256
  l2norm_rows_bf16<<<rows / 8, 256, 0, stream>>>(q_post, q_bf, rows, 0.0625f);
  l2norm_rows_bf16<<<rows / 8, 256, 0, stream>>>(k_post, k_bf, rows, 1.0f);

  const size_t scanLds =
      (size_t)(256 * LDS_S + 3 * 256 * LDT + 64 * LDS_S + 64 * LDT) * sizeof(unsigned short) +
      (size_t)(2 * 64 * LDA + 64) * sizeof(float);   // 323,840 B
  deltanet_scan<<<B * 4, 512, scanLds, stream>>>(q_bf, k_bf, v_bf, beta,
                                                 S0, o_raw, Sout, B, L);

  rmsnorm_rows_bf16<<<rows / 8, 256, 0, stream>>>(o_raw, nw, o_nrm, rows);

  gemm_bf16_wmma<<<ggrid, gblk, 0, stream>>>(o_nrm, Wo_bf, out, (int)BL, HIDc, HIDc);
}
